// BaseAlbertCrf_40140764348983
// MI455X (gfx1250) — compile-verified, matrix operand order checked by reading, not run on hardware
//
#include <hip/hip_runtime.h>

#define B_ 64
#define S_ 2048
#define L_ 128

#define LOG2E_F 1.44269504088896340736f
#define LN2_F   0.69314718055994530942f

typedef __attribute__((ext_vector_type(16))) _Float16 v16h;
typedef __attribute__((ext_vector_type(8)))  float    v8f;

// K index mapping for 16-bit WMMA A/B fragments (16x32 / 32x16), per ISA 7.12.2:
// lanes 0-15 hold K=0..7 (e<8) and K=16..23 (e>=8); lanes 16-31 hold K=8..15 / 24..31.
__device__ __forceinline__ int kmap(int e, int kbase) {
  return (e < 8) ? (kbase + e) : (16 + kbase + (e - 8));
}

__device__ __forceinline__ float grp16_max(float x) {
#pragma unroll
  for (int off = 1; off < 16; off <<= 1) x = fmaxf(x, __shfl_xor(x, off, 16));
  return x;
}
__device__ __forceinline__ float grp16_sum(float x) {
#pragma unroll
  for (int off = 1; off < 16; off <<= 1) x += __shfl_xor(x, off, 16);
  return x;
}

// Fast transcendentals: arguments are well-scaled by construction (rescaled
// exponents <= 0; log args in [~1e-2, ~1e3]), so raw v_exp_f32/v_log_f32 are safe.
__device__ __forceinline__ float fast_exp(float x) {
  return __builtin_amdgcn_exp2f(x * LOG2E_F);
}
__device__ __forceinline__ float fast_log(float x) {
  return __builtin_amdgcn_logf(x) * LN2_F;
}

// One block = 16 batch rows, 8 waves; wave w owns columns [16w, 16w+16).
// alpha kept in registers (float alf[8], WMMA C/D layout), E=exp(T) kept as
// persistent per-wave B fragments (4 x v16h = 32 VGPRs) for all 2048 steps.
__global__ __launch_bounds__(256) void crf_forward_kernel(
    const float* __restrict__ emissions, const int* __restrict__ mask,
    const float* __restrict__ trans, const float* __restrict__ startT,
    const float* __restrict__ endT, float* __restrict__ logZ)
{
  __shared__ _Float16 sP[16][L_];   // rescaled probs: A operand for next GEMM
  __shared__ float swmax[8][16];    // per-wave partial row maxes
  __shared__ float swsum[8][16];    // final logsumexp partial sums
  __shared__ int   smask[16];       // per-row mask bit for current step

  const int tid    = threadIdx.x;
  const int w      = tid >> 5;         // wave id -> N tile
  const int ln     = tid & 31;
  const int lane16 = ln & 15;
  const int half8  = (ln >> 4) * 8;    // row offset for C/D layout
  const int kbase  = (ln >> 4) * 8;    // K offset for A/B fragments
  const int col    = w * 16 + lane16;  // global column 0..127
  const int b0     = blockIdx.x * 16;

  // Persistent B fragments: E[k][col] = exp(transitions[k][col]), f16.
  v16h Bfr[4];
#pragma unroll
  for (int f = 0; f < 4; ++f) {
    _Float16* bp = reinterpret_cast<_Float16*>(&Bfr[f]);
#pragma unroll
    for (int e = 0; e < 16; ++e) {
      int k = 32 * f + kmap(e, kbase);
      bp[e] = (_Float16)fast_exp(trans[k * L_ + col]);
    }
  }

  // alpha_0 = start + emissions[:,0,:]
  float alf[8];
#pragma unroll
  for (int v = 0; v < 8; ++v) {
    int row = v + half8;
    alf[v] = startT[col] + emissions[((long)(b0 + row) * S_) * L_ + col];
  }

  for (int t = 1; t < S_; ++t) {
    // Phase A: issue this step's emission loads (non-temporal, consumed after
    // bar1) and next step's prefetch; exchange per-wave partial row maxes.
    float em[8];
#pragma unroll
    for (int v = 0; v < 8; ++v) {
      int row = v + half8;
      em[v] = __builtin_nontemporal_load(
          &emissions[(((long)(b0 + row)) * S_ + t) * L_ + col]);
    }
    if (t + 1 < S_)
      __builtin_prefetch(&emissions[(((long)(b0 + half8)) * S_ + t + 1) * L_ + col], 0, 0);
#pragma unroll
    for (int v = 0; v < 8; ++v) {
      float x = grp16_max(alf[v]);
      if (lane16 == 0) swmax[w][v + half8] = x;
    }
    __syncthreads();   // bar1: also separates step-t smask/sP reads from t+1 writes

    // Phase B: full row max via one-row-per-lane reduce + lane broadcast;
    // publish P = exp(alpha - rowmax) as f16 A-operand; publish step mask.
    float rfull = swmax[0][lane16];
#pragma unroll
    for (int w2 = 1; w2 < 8; ++w2) rfull = fmaxf(rfull, swmax[w2][lane16]);
    float rmax[8], base[8];
#pragma unroll
    for (int v = 0; v < 8; ++v) {
      rmax[v] = __shfl(rfull, v + half8, 32);
      int row = v + half8;
      sP[row][col] = (_Float16)fast_exp(alf[v] - rmax[v]);
      base[v] = rmax[v] + em[v];   // folded into post-WMMA fmac
    }
    if (w == 0 && ln < 16) smask[ln] = mask[(long)(b0 + ln) * S_ + t];
    __syncthreads();   // bar2

    // Phase C: preload all 4 A fragments (pipelined ds_load_b128s), then the
    // WMMA chain; nxt = (rowmax + emis) + log(P @ E); masked per-row update.
    v16h afr[4];
#pragma unroll
    for (int f = 0; f < 4; ++f) {
      unsigned int* ap = reinterpret_cast<unsigned int*>(&afr[f]);
#pragma unroll
      for (int i = 0; i < 8; ++i) {
        int k = 32 * f + kmap(2 * i, kbase);
        ap[i] = *reinterpret_cast<const unsigned int*>(&sP[lane16][k]);
      }
    }
    int mrow[8];
#pragma unroll
    for (int v = 0; v < 8; ++v) mrow[v] = smask[v + half8];

    v8f acc = {};
#pragma unroll
    for (int f = 0; f < 4; ++f)
      acc = __builtin_amdgcn_wmma_f32_16x16x32_f16(
          false, afr[f], false, Bfr[f], (short)0, acc, false, false);

#pragma unroll
    for (int v = 0; v < 8; ++v) {
      float nxt = base[v] + fast_log(acc[v]);
      if (mrow[v]) alf[v] = nxt;
    }
  }

  // Final: logZ[b] = logsumexp_col(alpha + end_transitions)
  float ev[8];
#pragma unroll
  for (int v = 0; v < 8; ++v) ev[v] = alf[v] + endT[col];
#pragma unroll
  for (int v = 0; v < 8; ++v) {
    float x = grp16_max(ev[v]);
    if (lane16 == 0) swmax[w][v + half8] = x;
  }
  __syncthreads();
#pragma unroll
  for (int v = 0; v < 8; ++v) {
    int row = v + half8;
    float rm = swmax[0][row];
#pragma unroll
    for (int w2 = 1; w2 < 8; ++w2) rm = fmaxf(rm, swmax[w2][row]);
    float s = grp16_sum(fast_exp(ev[v] - rm));
    if (lane16 == 0) swsum[w][row] = s;
  }
  __syncthreads();
  if (w == 0 && ln < 16) {
    int r = ln;
    float rm = swmax[0][r];
    float sum = swsum[0][r];
#pragma unroll
    for (int w2 = 1; w2 < 8; ++w2) { rm = fmaxf(rm, swmax[w2][r]); sum += swsum[w2][r]; }
    logZ[b0 + r] = rm + fast_log(sum);
  }
}

// Gold-path score: one block per batch element, threads stride over time.
__global__ __launch_bounds__(256) void crf_score_kernel(
    const float* __restrict__ emissions, const int* __restrict__ mask,
    const int* __restrict__ labels, const float* __restrict__ trans,
    const float* __restrict__ startT, const float* __restrict__ endT,
    float* __restrict__ score)
{
  const int b = blockIdx.x, tid = threadIdx.x;
  __shared__ float sred[256];
  __shared__ int scnt[256];
  float s = 0.0f;
  int cnt = 0;
  for (int t = tid; t < S_; t += 256) {
    int m = mask[(long)b * S_ + t];
    cnt += m;
    if (t >= 1 && m) {
      int lt = labels[(long)b * S_ + t];
      int lp = labels[(long)b * S_ + t - 1];
      s += __builtin_nontemporal_load(&emissions[((long)b * S_ + t) * L_ + lt]) +
           trans[lp * L_ + lt];
    }
  }
  sred[tid] = s; scnt[tid] = cnt;
  __syncthreads();
  for (int st = 128; st > 0; st >>= 1) {
    if (tid < st) { sred[tid] += sred[tid + st]; scnt[tid] += scnt[tid + st]; }
    __syncthreads();
  }
  if (tid == 0) {
    int l0 = labels[(long)b * S_];
    int last = labels[(long)b * S_ + scnt[0] - 1];
    score[b] = sred[0] + startT[l0] + emissions[(long)b * S_ * L_ + l0] + endT[last];
  }
}

__global__ void crf_final_kernel(const float* __restrict__ logZ,
                                 const float* __restrict__ score,
                                 float* __restrict__ out)
{
  __shared__ float sr[64];
  int tid = threadIdx.x;
  sr[tid] = logZ[tid] - score[tid];
  __syncthreads();
  for (int st = 32; st > 0; st >>= 1) {
    if (tid < st) sr[tid] += sr[tid + st];
    __syncthreads();
  }
  if (tid == 0) out[0] = sr[0] / (float)B_;
}

extern "C" void kernel_launch(void* const* d_in, const int* in_sizes, int n_in,
                              void* d_out, int out_size, void* d_ws, size_t ws_size,
                              hipStream_t stream) {
  const float* emissions = (const float*)d_in[0];
  const int*   attn      = (const int*)d_in[1];
  const int*   labels    = (const int*)d_in[2];
  const float* trans     = (const float*)d_in[3];
  const float* startT    = (const float*)d_in[4];
  const float* endT      = (const float*)d_in[5];
  float* out   = (float*)d_out;
  float* logZ  = (float*)d_ws;        // 64 floats
  float* score = logZ + B_;           // 64 floats

  crf_forward_kernel<<<dim3(B_ / 16), dim3(256), 0, stream>>>(
      emissions, attn, trans, startT, endT, logZ);
  crf_score_kernel<<<dim3(B_), dim3(256), 0, stream>>>(
      emissions, attn, labels, trans, startT, endT, score);
  crf_final_kernel<<<dim3(1), dim3(64), 0, stream>>>(logZ, score, out);
}
